// ExportableGlobalTransformerCore_30442728194393
// MI455X (gfx1250) — matrix operand order repaired and checked by reading, not compiled, not run on hardware
//
#include <hip/hip_runtime.h>
#include <math.h>

#define L_ 4
#define B_ 2
#define S_ 1024
#define D_ 1024
#define H_ 16
#define DH_ 64
#define F_ 4096

typedef __bf16 bf16;
typedef __attribute__((ext_vector_type(16))) __bf16 v16bf;
typedef __attribute__((ext_vector_type(8)))  float  v8f;

__device__ __forceinline__ bf16 tobf(float f) { return (bf16)f; }

// truncate generic pointer -> LDS segment byte offset (AS3 value)
__device__ __forceinline__ unsigned ldsaddr(const void* p) {
    return (unsigned)(size_t)p;
}
// async 16B copy global -> LDS (ASYNCcnt tracked)
__device__ __forceinline__ void async_b128(unsigned lds, const void* gsrc) {
    asm volatile("global_load_async_to_lds_b128 %0, %1, off"
                 :: "v"(lds), "v"(gsrc) : "memory");
}
__device__ __forceinline__ void wait_async0() {
    asm volatile("s_wait_asynccnt 0x0" ::: "memory");
}

// ---------------------------------------------------------------------------
// pos_ids = cumsum(mask) - 1 (clamped to 0 where mask==0)
// ---------------------------------------------------------------------------
__global__ void pos_kernel(const unsigned char* __restrict__ amask, int* __restrict__ pos)
{
    int idx = blockIdx.x * blockDim.x + threadIdx.x;
    if (idx >= B_ * S_) return;
    int b = idx / S_, s = idx % S_;
    int p = 0;
    for (int i = 0; i <= s; ++i) p += amask[b * S_ + i] ? 1 : 0;
    pos[idx] = amask[idx] ? (p - 1) : 0;
}

// h = inputs_embeds * mask
__global__ void init_kernel(const float* __restrict__ x, const unsigned char* __restrict__ m,
                            float* __restrict__ h)
{
    int i = blockIdx.x * blockDim.x + threadIdx.x;
    if (i >= B_ * S_ * D_) return;
    int row = i / D_;
    h[i] = x[i] * (m[row] ? 1.0f : 0.0f);
}

// ---------------------------------------------------------------------------
// Tiled transpose + fp32->bf16 convert:  W[K][N] -> Wt[N][K]
// ---------------------------------------------------------------------------
__global__ __launch_bounds__(256)
void convt_kernel(const float* __restrict__ W, bf16* __restrict__ Wt, int K, int N)
{
    __shared__ bf16 sh[32][33];
    const int n0 = blockIdx.x * 32;
    const int k0 = blockIdx.y * 32;
    const int tx = threadIdx.x & 31;
    const int ty = threadIdx.x >> 5;            // 8 rows of the tile per pass
    for (int r = ty; r < 32; r += 8)
        sh[r][tx] = tobf(W[(size_t)(k0 + r) * N + n0 + tx]);
    __syncthreads();
    for (int r = ty; r < 32; r += 8)
        Wt[(size_t)(n0 + r) * K + k0 + tx] = sh[tx][r];
}

// ---------------------------------------------------------------------------
// LayerNorm (block per row); writes bf16 (GEMM A input) or fp32 (final out)
// ---------------------------------------------------------------------------
__global__ __launch_bounds__(256)
void ln_kernel(const float* __restrict__ x, const float* __restrict__ g,
               const float* __restrict__ bvec, bf16* __restrict__ outb,
               float* __restrict__ outf, const unsigned char* __restrict__ rowmask)
{
    __shared__ float red[256];
    const int row = blockIdx.x;
    const float* xr = x + (size_t)row * D_;

    float s = 0.f;
    for (int i = threadIdx.x; i < D_; i += 256) s += xr[i];
    red[threadIdx.x] = s; __syncthreads();
    for (int st = 128; st > 0; st >>= 1) {
        if (threadIdx.x < st) red[threadIdx.x] += red[threadIdx.x + st];
        __syncthreads();
    }
    const float mu = red[0] * (1.0f / D_);
    __syncthreads();

    float s2 = 0.f;
    for (int i = threadIdx.x; i < D_; i += 256) { float d = xr[i] - mu; s2 += d * d; }
    red[threadIdx.x] = s2; __syncthreads();
    for (int st = 128; st > 0; st >>= 1) {
        if (threadIdx.x < st) red[threadIdx.x] += red[threadIdx.x + st];
        __syncthreads();
    }
    const float rstd = rsqrtf(red[0] * (1.0f / D_) + 1e-5f);
    const float mk = rowmask ? (rowmask[row] ? 1.0f : 0.0f) : 1.0f;

    for (int i = threadIdx.x; i < D_; i += 256) {
        float v = ((xr[i] - mu) * rstd * g[i] + bvec[i]) * mk;
        if (outb) outb[(size_t)row * D_ + i] = tobf(v);
        else      outf[(size_t)row * D_ + i] = v;
    }
}

// ---------------------------------------------------------------------------
// WMMA bf16 GEMM: out[M,N] = epi( A[M,K](bf16) @ Wt[N,K]^T(bf16) + bias (+res) )
// block tile 64x128, 8 waves, wave tile 16x64; LDS tiles staged with
// global_load_async_to_lds_b128 (ASYNCcnt), next tile prefetched.
// ---------------------------------------------------------------------------
__global__ __launch_bounds__(256)
void gemm_kernel(const bf16* __restrict__ A, const bf16* __restrict__ Wt,
                 const float* __restrict__ bias, const float* __restrict__ res,
                 const unsigned char* __restrict__ rowmask,
                 float* __restrict__ outf, bf16* __restrict__ outb,
                 int M, int N, int K, int gelu)
{
    __shared__ __align__(16) bf16 As[64][32];     // [m][k]
    __shared__ __align__(16) bf16 Bs[128][32];    // [n][k]

    const int tid  = threadIdx.x;
    const int lane = tid & 31;
    const int wid  = tid >> 5;
    const int half = lane >> 4;
    const int ln16 = lane & 15;
    const int wm   = wid & 3;
    const int wn   = wid >> 2;
    const int mblk = blockIdx.y * 64;
    const int nblk = blockIdx.x * 128;

    // per-thread staging chunks (16B each, all 16B-aligned)
    const int arow = tid >> 2, ak = (tid & 3) * 8;          // A: 1 chunk
    const int bn0  = tid >> 1, bk0 = (tid & 1) * 16;        // B: 2 chunks (k, k+8)

    v8f z = {};
    v8f acc[4];
    for (int g = 0; g < 4; ++g) acc[g] = z;

    const int kt = K >> 5;
    for (int kb = 0; kb < kt; ++kb) {
        __syncthreads();
        // async stage A tile 64x32 (4KB): one b128 per thread
        async_b128(ldsaddr(&As[arow][ak]),
                   A + (size_t)(mblk + arow) * K + kb * 32 + ak);
        // async stage B tile 128x32 (8KB): two b128 per thread
        async_b128(ldsaddr(&Bs[bn0][bk0]),
                   Wt + (size_t)(nblk + bn0) * K + kb * 32 + bk0);
        async_b128(ldsaddr(&Bs[bn0][bk0 + 8]),
                   Wt + (size_t)(nblk + bn0) * K + kb * 32 + bk0 + 8);
        if (kb + 1 < kt) {   // prefetch next k-tile (global_prefetch_b8)
            __builtin_prefetch(A  + (size_t)(mblk + arow) * K + (kb + 1) * 32 + ak, 0, 1);
            __builtin_prefetch(Wt + (size_t)(nblk + bn0) * K + (kb + 1) * 32 + bk0, 0, 1);
        }
        wait_async0();
        __syncthreads();

        v16bf aF;
        {
            int ar = wm * 16 + ln16;
            #pragma unroll
            for (int i = 0; i < 16; ++i) {
                int kk = half * 16 + i;
                aF[i] = As[ar][kk];
            }
        }
        #pragma unroll
        for (int g = 0; g < 4; ++g) {
            v16bf bF;
            int ncol = wn * 64 + g * 16 + ln16;
            #pragma unroll
            for (int i = 0; i < 16; ++i) bF[i] = Bs[ncol][half * 16 + i];
            acc[g] = __builtin_amdgcn_wmma_f32_16x16x32_bf16(
                false, aF, false, bF, (short)0, acc[g], false, false);
        }
    }

    // epilogue: bias -> gelu(sigmoid form, branch-free) -> +res -> *rowmask
    #pragma unroll
    for (int g = 0; g < 4; ++g) {
        #pragma unroll
        for (int r = 0; r < 8; ++r) {
            int m = mblk + wm * 16 + half * 8 + r;
            int n = nblk + wn * 64 + g * 16 + ln16;
            float v = acc[g][r] + bias[n];
            if (gelu) {
                float u2 = 1.5957691216057308f * (v + 0.044715f * v * v * v);
                v = v / (1.0f + __expf(-u2));
            }
            if (res)     v += res[(size_t)m * N + n];
            if (rowmask) v *= rowmask[m] ? 1.0f : 0.0f;
            if (outb) outb[(size_t)m * N + n] = tobf(v);
            else      outf[(size_t)m * N + n] = v;
        }
    }
}

// ---------------------------------------------------------------------------
// RoPE + head relayout: qkv[B,S,3D] -> q/k/v [B,H,S,DH] (fp32)
// ---------------------------------------------------------------------------
__global__ void rope_kernel(const float* __restrict__ qkv, const int* __restrict__ pos,
                            float* __restrict__ q, float* __restrict__ k, float* __restrict__ v)
{
    int idx = blockIdx.x * blockDim.x + threadIdx.x;       // B*S*H*(DH/2)
    if (idx >= B_ * S_ * H_ * (DH_ / 2)) return;
    int i = idx & 31;
    int h = (idx >> 5) & (H_ - 1);
    int s = (idx >> 9) & (S_ - 1);
    int b = idx >> 19;

    float p   = (float)pos[b * S_ + s];
    float inv = __expf(-((2.0f * (float)i) / (float)DH_) * 9.210340371976184f); // ln(10000)
    float ang = p * inv;
    float c = __cosf(ang), sn = __sinf(ang);

    size_t base = ((size_t)(b * S_ + s)) * (3 * D_) + h * DH_;
    float q0 = qkv[base + 2 * i],          q1 = qkv[base + 2 * i + 1];
    float k0 = qkv[base + D_ + 2 * i],     k1 = qkv[base + D_ + 2 * i + 1];
    float v0 = qkv[base + 2 * D_ + 2 * i], v1 = qkv[base + 2 * D_ + 2 * i + 1];

    size_t ob = (((size_t)(b * H_ + h)) * S_ + s) * DH_ + 2 * i;
    q[ob] = q0 * c - q1 * sn;  q[ob + 1] = q1 * c + q0 * sn;
    k[ob] = k0 * c - k1 * sn;  k[ob + 1] = k1 * c + k0 * sn;
    v[ob] = v0;                v[ob + 1] = v1;
}

// ---------------------------------------------------------------------------
// Flash attention: block = (b, h, 128 query rows); 8 waves x 16-row q tiles.
// WMMA bf16 for Q@K^T and P@V; online softmax; bf16 output.
// ---------------------------------------------------------------------------
__global__ __launch_bounds__(256)
void attn_kernel(const float* __restrict__ q, const float* __restrict__ k,
                 const float* __restrict__ v, const unsigned char* __restrict__ amask,
                 bf16* __restrict__ out)
{
    __shared__ bf16 Ks[32][64];        // [key][dh]
    __shared__ bf16 Vt[64][32];        // [dh][key]
    __shared__ bf16 Pl[8][16][32];     // per-wave P tile [m][key]

    const int QB = S_ / 128;
    const int qb = blockIdx.x % QB;
    const int h  = (blockIdx.x / QB) % H_;
    const int b  = blockIdx.x / (QB * H_);

    const int tid  = threadIdx.x;
    const int lane = tid & 31;
    const int wid  = tid >> 5;
    const int half = lane >> 4;
    const int ln16 = lane & 15;
    const int qblk = qb * 128;
    const int qw   = qblk + wid * 16;
    const size_t headOff = ((size_t)(b * H_ + h)) * S_ * DH_;

    v16bf aQ[2];
    {
        const float* qp = q + headOff + (size_t)(qw + ln16) * DH_;
        #pragma unroll
        for (int j = 0; j < 2; ++j)
            #pragma unroll
            for (int i = 0; i < 16; ++i) {
                int dh = 32 * j + half * 16 + i;
                aQ[j][i] = tobf(qp[dh]);
            }
    }

    v8f z = {};
    v8f acc[4];
    for (int g = 0; g < 4; ++g) acc[g] = z;
    float mrow[8], lrow[8];
    #pragma unroll
    for (int r = 0; r < 8; ++r) { mrow[r] = -1e30f; lrow[r] = 0.f; }

    const int nkt = (qblk + 128) >> 5;
    for (int kt2 = 0; kt2 < nkt; ++kt2) {
        const int kbase = kt2 * 32;
        __syncthreads();
        {
            int key = tid >> 3;
            int d0  = (tid & 7) * 8;
            const float* kp = k + headOff + (size_t)(kbase + key) * DH_ + d0;
            const float* vp = v + headOff + (size_t)(kbase + key) * DH_ + d0;
            #pragma unroll
            for (int i = 0; i < 8; ++i) {
                Ks[key][d0 + i] = tobf(kp[i]);
                Vt[d0 + i][key] = tobf(vp[i]);
            }
        }
        __syncthreads();

        const bool active = (kbase < qw + 16);  // wave-uniform
        if (active) {
            v16bf bK;
            v8f s0 = z, s1 = z;
            #pragma unroll
            for (int j = 0; j < 2; ++j) {
                #pragma unroll
                for (int i = 0; i < 16; ++i) bK[i] = Ks[ln16][32 * j + half * 16 + i];
                s0 = __builtin_amdgcn_wmma_f32_16x16x32_bf16(
                    false, aQ[j], false, bK, (short)0, s0, false, false);
            }
            #pragma unroll
            for (int j = 0; j < 2; ++j) {
                #pragma unroll
                for (int i = 0; i < 16; ++i) bK[i] = Ks[16 + ln16][32 * j + half * 16 + i];
                s1 = __builtin_amdgcn_wmma_f32_16x16x32_bf16(
                    false, aQ[j], false, bK, (short)0, s1, false, false);
            }

            const int j0 = kbase + ln16;
            const int j1 = kbase + 16 + ln16;
            const bool km0 = amask[b * S_ + j0] != 0;
            const bool km1 = amask[b * S_ + j1] != 0;

            #pragma unroll
            for (int r = 0; r < 8; ++r) {
                const int qrow = qw + half * 8 + r;
                float sc0 = s0[r] * 0.125f;
                float sc1 = s1[r] * 0.125f;
                if (j0 > qrow || !km0) sc0 = -1e30f;
                if (j1 > qrow || !km1) sc1 = -1e30f;

                float rm = fmaxf(sc0, sc1);
                #pragma unroll
                for (int msk = 1; msk < 16; msk <<= 1)
                    rm = fmaxf(rm, __shfl_xor(rm, msk, 32));
                const float nm   = fmaxf(mrow[r], rm);
                const float corr = __expf(mrow[r] - nm);
                mrow[r] = nm;

                const float e0 = __expf(sc0 - nm);
                const float e1 = __expf(sc1 - nm);
                float rs = e0 + e1;
                #pragma unroll
                for (int msk = 1; msk < 16; msk <<= 1)
                    rs += __shfl_xor(rs, msk, 32);
                lrow[r] = lrow[r] * corr + rs;
                #pragma unroll
                for (int g = 0; g < 4; ++g) acc[g][r] *= corr;

                Pl[wid][half * 8 + r][ln16]      = tobf(e0);
                Pl[wid][half * 8 + r][16 + ln16] = tobf(e1);
            }
        }
        __syncthreads();
        if (active) {
            v16bf aP;
            #pragma unroll
            for (int i = 0; i < 16; ++i) {
                int kk = half * 16 + i;
                aP[i] = Pl[wid][ln16][kk];
            }
            #pragma unroll
            for (int g = 0; g < 4; ++g) {
                v16bf bV;
                #pragma unroll
                for (int i = 0; i < 16; ++i) bV[i] = Vt[g * 16 + ln16][half * 16 + i];
                acc[g] = __builtin_amdgcn_wmma_f32_16x16x32_bf16(
                    false, aP, false, bV, (short)0, acc[g], false, false);
            }
        }
    }

    #pragma unroll
    for (int r = 0; r < 8; ++r) {
        const float inv  = 1.0f / fmaxf(lrow[r], 1e-20f);
        const int   qrow = qw + half * 8 + r;
        bf16* op = out + ((size_t)(b * S_ + qrow)) * D_ + h * DH_;
        #pragma unroll
        for (int g = 0; g < 4; ++g) op[g * 16 + ln16] = tobf(acc[g][r] * inv);
    }
}

// ---------------------------------------------------------------------------
extern "C" void kernel_launch(void* const* d_in, const int* in_sizes, int n_in,
                              void* d_out, int out_size, void* d_ws, size_t ws_size,
                              hipStream_t stream)
{
    (void)in_sizes; (void)n_in; (void)out_size; (void)ws_size;
    const float*         inp   = (const float*)d_in[0];
    const unsigned char* amask = (const unsigned char*)d_in[1];
    const float* w_qkv  = (const float*)d_in[2];
    const float* b_qkv  = (const float*)d_in[3];
    const float* w_proj = (const float*)d_in[4];
    const float* b_proj = (const float*)d_in[5];
    const float* w_fc   = (const float*)d_in[6];
    const float* b_fc   = (const float*)d_in[7];
    const float* w_out  = (const float*)d_in[8];
    const float* b_out  = (const float*)d_in[9];
    const float* ln1_g  = (const float*)d_in[10];
    const float* ln1_b  = (const float*)d_in[11];
    const float* ln2_g  = (const float*)d_in[12];
    const float* ln2_b  = (const float*)d_in[13];
    const float* lnf_g  = (const float*)d_in[14];
    const float* lnf_b  = (const float*)d_in[15];

    char* ws = (char*)d_ws;
    size_t off = 0;
    auto carve = [&](size_t bytes) -> char* {
        char* p = ws + off;
        off += (bytes + 255) & ~(size_t)255;
        return p;
    };
    int*   pos  = (int*)  carve((size_t)B_ * S_ * sizeof(int));
    float* h    = (float*)carve((size_t)B_ * S_ * D_ * 4);
    bf16*  tmpb = (bf16*) carve((size_t)B_ * S_ * D_ * 2);   // xln1 / attn / xln2 (bf16)
    float* big  = (float*)carve((size_t)B_ * S_ * F_ * 4);   // qkv fp32 / ffn mid bf16
    bf16*  midb = (bf16*) big;
    bf16*  wt   = (bf16*) carve((size_t)D_ * F_ * 2);        // transposed bf16 weights
    float* qh   = (float*)carve((size_t)B_ * H_ * S_ * DH_ * 4);
    float* kh   = (float*)carve((size_t)B_ * H_ * S_ * DH_ * 4);
    float* vh   = (float*)carve((size_t)B_ * H_ * S_ * DH_ * 4);

    const int M = B_ * S_;

    pos_kernel<<<(M + 255) / 256, 256, 0, stream>>>(amask, pos);
    init_kernel<<<(M * D_ + 255) / 256, 256, 0, stream>>>(inp, amask, h);

    for (int l = 0; l < L_; ++l) {
        // x = LN1(h)  (bf16)
        ln_kernel<<<M, 256, 0, stream>>>(h, ln1_g + l * D_, ln1_b + l * D_, tmpb, nullptr, nullptr);
        // qkv = x @ w_qkv + b   (fp32 out for RoPE)
        convt_kernel<<<dim3((3 * D_) / 32, D_ / 32), 256, 0, stream>>>(
            w_qkv + (size_t)l * D_ * 3 * D_, wt, D_, 3 * D_);
        gemm_kernel<<<dim3((3 * D_) / 128, M / 64), 256, 0, stream>>>(
            tmpb, wt, b_qkv + (size_t)l * 3 * D_, nullptr, nullptr,
            big, nullptr, M, 3 * D_, D_, 0);
        // rope + head relayout
        rope_kernel<<<(B_ * S_ * H_ * (DH_ / 2) + 255) / 256, 256, 0, stream>>>(big, pos, qh, kh, vh);
        // attention -> tmpb [B,S,D] bf16
        attn_kernel<<<B_ * H_ * (S_ / 128), 256, 0, stream>>>(qh, kh, vh, amask, tmpb);
        // h = h + attn @ w_proj + b
        convt_kernel<<<dim3(D_ / 32, D_ / 32), 256, 0, stream>>>(
            w_proj + (size_t)l * D_ * D_, wt, D_, D_);
        gemm_kernel<<<dim3(D_ / 128, M / 64), 256, 0, stream>>>(
            tmpb, wt, b_proj + (size_t)l * D_, h, nullptr, h, nullptr, M, D_, D_, 0);
        // y = LN2(h) (bf16)
        ln_kernel<<<M, 256, 0, stream>>>(h, ln2_g + l * D_, ln2_b + l * D_, tmpb, nullptr, nullptr);
        // mid = gelu(y @ w_fc + b)  (bf16 out)
        convt_kernel<<<dim3(F_ / 32, D_ / 32), 256, 0, stream>>>(
            w_fc + (size_t)l * D_ * F_, wt, D_, F_);
        gemm_kernel<<<dim3(F_ / 128, M / 64), 256, 0, stream>>>(
            tmpb, wt, b_fc + (size_t)l * F_, nullptr, nullptr,
            nullptr, midb, M, F_, D_, 1);
        // h = (h + mid @ w_out + b) * mask
        convt_kernel<<<dim3(D_ / 32, F_ / 32), 256, 0, stream>>>(
            w_out + (size_t)l * F_ * D_, wt, F_, D_);
        gemm_kernel<<<dim3(D_ / 128, M / 64), 256, 0, stream>>>(
            midb, wt, b_out + (size_t)l * D_, h, amask, h, nullptr, M, D_, F_, 0);
    }
    // out = LN_f(h) * mask  (fp32)
    ln_kernel<<<M, 256, 0, stream>>>(h, lnf_g, lnf_b, nullptr, (float*)d_out, amask);
}
